// TripleAttention_21809843929840
// MI455X (gfx1250) — matrix operand order repaired and checked by reading, not compiled, not run on hardware
//
#include <hip/hip_runtime.h>

typedef _Float16 f16;
typedef __attribute__((ext_vector_type(16))) _Float16 v16h;
typedef __attribute__((ext_vector_type(8)))  _Float16 v8h;
typedef __attribute__((ext_vector_type(8)))  float    v8f;
typedef int v4i_ __attribute__((vector_size(16)));

#define N      384
#define NT     24           // N/16
#define CC     64
#define SCALE  0.125f       // D^-0.5 = 1/8
#define LSTR   400          // padded LDS row stride in halves (16 rows per strip)

// ---------- WMMA helper ----------
__device__ __forceinline__ v8f wmma_f16(v16h a, v16h b, v8f c) {
  return __builtin_amdgcn_wmma_f32_16x16x32_f16(false, a, false, b, (short)0, c, false, false);
}

__device__ __forceinline__ v16h cat8(v8h lo, v8h hi) {
  return __builtin_shufflevector(lo, hi, 0,1,2,3,4,5,6,7,8,9,10,11,12,13,14,15);
}

// A-frag 16x32 f16 from row-major memory (global): lane holds row (lane&15);
// halves 0..7 at K = kb, halves 8..15 at K = kb+16, kb = (lane>>4)*8.
__device__ __forceinline__ v16h load_afrag(const f16* __restrict__ arow, int j0, int lane) {
  int kb = (lane >> 4) * 8;
  v8h lo = *(const v8h*)(arow + j0 + kb);
  v8h hi = *(const v8h*)(arow + j0 + kb + 16);
  return cat8(lo, hi);
}

// ---------- async global->LDS copy (CDNA5 ASYNCcnt path, with fallback) ----------
#if __has_builtin(__builtin_amdgcn_global_load_async_to_lds_b128)
#define HAS_ASYNC_LDS 1
#else
#define HAS_ASYNC_LDS 0
#endif

__device__ __forceinline__ void async_copy16(const f16* gsrc, f16* lds_dst) {
#if HAS_ASYNC_LDS
  __builtin_amdgcn_global_load_async_to_lds_b128(
      (__attribute__((address_space(1))) v4i_*)(void*)const_cast<f16*>(gsrc),
      (__attribute__((address_space(3))) v4i_*)(void*)lds_dst,
      0, 0);
#else
  *(v8h*)lds_dst = *(const v8h*)gsrc;
#endif
}

__device__ __forceinline__ void async_wait() {
#if HAS_ASYNC_LDS
#if __has_builtin(__builtin_amdgcn_s_wait_asynccnt)
  __builtin_amdgcn_s_wait_asynccnt(0);
#else
  asm volatile("s_wait_asynccnt 0x0" ::: "memory");
#endif
#endif
}

// copy a contiguous 16x384-half strip (6144 halves) into a padded LDS strip [16][LSTR]
__device__ __forceinline__ void copy_strip(f16* lds_base, const f16* __restrict__ gbase, int tid) {
  #pragma unroll
  for (int c = 0; c < 3; ++c) {
    int lin = (tid + c * 256) * 8;       // 8-half chunks, rows of 384 divide evenly
    int r = lin / 384, j = lin % 384;
    async_copy16(gbase + lin, lds_base + r * LSTR + j);
  }
}

// ---------- stage 0: f32 -> f16 conversion ----------
__global__ void k_cvt(const float* __restrict__ x, const float* __restrict__ w,
                      f16* __restrict__ xh, f16* __restrict__ wh) {
  int idx = blockIdx.x * blockDim.x + threadIdx.x;
  const int nx = N * CC;
  const int nw = 8 * CC * CC;
  if (idx < nx) xh[idx] = (f16)x[idx];
  else if (idx < nx + nw) wh[idx - nx] = (f16)w[idx - nx];
}

// ---------- stage 1: projection  w8h[384][512] = xh @ wh^T ----------
__global__ void k_proj(const f16* __restrict__ xh, const f16* __restrict__ wh,
                       f16* __restrict__ w8h) {
  int it = blockIdx.x, jt = blockIdx.y;
  int lane = threadIdx.x;
  int m = lane & 15, g = lane >> 4;
  const f16* arow = xh + (it * 16 + m) * CC;
  const f16* brow = wh + (jt * 16 + m) * CC;
  v8f acc = {};
  #pragma unroll
  for (int ks = 0; ks < 2; ++ks) {
    v16h a = load_afrag(arow, ks * 32, lane);
    v16h b = cat8(*(const v8h*)(brow + ks * 32 + g * 8), *(const v8h*)(brow + ks * 32 + g * 8 + 16));
    acc = wmma_f16(a, b, acc);
  }
  int i0 = it * 16 + g * 8;
  int j  = jt * 16 + m;
  #pragma unroll
  for (int r = 0; r < 8; ++r) w8h[(i0 + r) * 512 + j] = (f16)acc[r];
}

// ---------- stage 2: score matrices ----------
__global__ void k_scores(const f16* __restrict__ w8h, float* __restrict__ S) {
  int it = blockIdx.x, jt = blockIdx.y, s = blockIdx.z;
  int lane = threadIdx.x;
  int m = lane & 15, g = lane >> 4;
  const f16* arow = w8h + (it * 16 + m) * 512 + (2 * s + 0) * 64;
  const f16* brow = w8h + (jt * 16 + m) * 512 + (2 * s + 1) * 64;
  v8f acc = {};
  #pragma unroll
  for (int ks = 0; ks < 2; ++ks) {
    v16h a = load_afrag(arow, ks * 32, lane);
    v16h b = cat8(*(const v8h*)(brow + ks * 32 + g * 8), *(const v8h*)(brow + ks * 32 + g * 8 + 16));
    acc = wmma_f16(a, b, acc);
  }
  float* out = S + (size_t)s * N * N;
  int i0 = it * 16 + g * 8;
  int j  = jt * 16 + m;
  #pragma unroll
  for (int r = 0; r < 8; ++r) out[(i0 + r) * N + j] = acc[r] * SCALE;
}

// ---------- stage 3: per-matrix max ----------
__global__ void k_max3(const float* __restrict__ S, float* __restrict__ m3) {
  int s = blockIdx.x;
  const float* p = S + (size_t)s * N * N;
  float mx = -1e30f;
  for (int i = threadIdx.x; i < N * N; i += blockDim.x) mx = fmaxf(mx, p[i]);
  __shared__ float red[256];
  red[threadIdx.x] = mx;
  __syncthreads();
  for (int off = 128; off > 0; off >>= 1) {
    if ((int)threadIdx.x < off) red[threadIdx.x] = fmaxf(red[threadIdx.x], red[threadIdx.x + off]);
    __syncthreads();
  }
  if (threadIdx.x == 0) m3[s] = red[0];
}

// ---------- stage 4: exponentials -> P, Q, R^T (f16) ----------
__global__ void k_exp(const float* __restrict__ S, const float* __restrict__ m3,
                      f16* __restrict__ Ph, f16* __restrict__ Qh, f16* __restrict__ Rth) {
  int idx = blockIdx.x * blockDim.x + threadIdx.x;
  if (idx >= 3 * N * N) return;
  int s = idx / (N * N);
  int t = idx % (N * N);
  float v = __expf(S[idx] - m3[s]);
  if (s == 0)      Ph[t] = (f16)v;
  else if (s == 1) Qh[t] = (f16)v;
  else { int j = t / N, k = t % N; Rth[k * N + j] = (f16)v; }
}

// ---------- stage 5: transpose v1,v2 ----------
__global__ void k_vt(const f16* __restrict__ w8h, f16* __restrict__ v1t, f16* __restrict__ v2t) {
  int idx = blockIdx.x * blockDim.x + threadIdx.x;
  if (idx >= 2 * CC * N) return;
  int sel = idx / (CC * N);
  int r   = idx % (CC * N);
  int dd = r / N, j = r % N;
  f16 val = w8h[j * 512 + 384 + sel * 64 + dd];
  (sel ? v2t : v1t)[dd * N + j] = val;
}

// ---------- stage 6: main contraction, LDS-staged ----------
// Block: 256 threads = 8 waves, one i-tile; wave w handles d-slot blockIdx.y*8+w.
// num[i,d] = sum_k Q[i,k]*v2[k,d] * ( (P . v1[d]) @ R )[i,k]      (slot < 64)
// den[i]   = sum_k Q[i,k]         * (  P          @ R )[i,k]      (slot == 64)
__global__ void __launch_bounds__(256)
k_main(const f16* __restrict__ Ph, const f16* __restrict__ Qh,
       const f16* __restrict__ Rth, const f16* __restrict__ v1t,
       const f16* __restrict__ v2t,
       float* __restrict__ num, float* __restrict__ den) {
  __shared__ f16 Plds[16 * LSTR];
  __shared__ f16 Rlds[2][16 * LSTR];

  int it   = blockIdx.x;
  int tid  = threadIdx.x;
  int lane = tid & 31;
  int wave = tid >> 5;
  int slot = blockIdx.y * 8 + wave;     // 0..71 (65..71 = redundant den, no store)
  int m = lane & 15, g = lane >> 4;
  bool isNum = (slot < 64);

  // prologue: stage P tile and R strip 0 into LDS
  copy_strip(Plds,    Ph  + (size_t)it * 16 * N, tid);
  copy_strip(Rlds[0], Rth, tid);
  async_wait();
  __syncthreads();

  // per-wave: build scaled A-fragments  As = P_frag (* v1[d]_frag)
  const f16* v1row = v1t + (isNum ? slot : 0) * N;
  const f16* v2row = v2t + (isNum ? slot : 0) * N;
  const f16* prow  = Plds + m * LSTR;
  v16h As[12];
  #pragma unroll
  for (int js = 0; js < 12; ++js) {
    int kb = (g << 3);
    int j0 = js * 32;
    v8h lo = *(const v8h*)(prow + j0 + kb);
    v8h hi = *(const v8h*)(prow + j0 + kb + 16);
    v16h pa = cat8(lo, hi);
    if (isNum) {
      v16h vf = load_afrag(v1row, j0, lane);
      pa = pa * vf;                       // packed f16
    }
    As[js] = pa;
  }

  float numacc[8] = {0.f,0.f,0.f,0.f,0.f,0.f,0.f,0.f};

  for (int kt = 0; kt < NT; ++kt) {
    // prefetch next strip into the other buffer (overlaps with WMMA below)
    if (kt + 1 < NT)
      copy_strip(Rlds[(kt + 1) & 1], Rth + (size_t)(kt + 1) * 16 * N, tid);

    int k = kt * 16 + m;                  // this lane's output column (global k)
    const f16* rrow = Rlds[kt & 1] + m * LSTR;
    v8f acc = {};
    #pragma unroll
    for (int js = 0; js < 12; ++js) {
      int jb = js * 32 + g * 8;
      v8h lo = *(const v8h*)(rrow + jb);
      v8h hi = *(const v8h*)(rrow + jb + 16);
      acc = wmma_f16(As[js], cat8(lo, hi), acc);
    }
    // epilogue: * Q[i,k] (* v2[k,d]), accumulate over k
    float v2v = isNum ? (float)v2row[k] : 1.0f;
    #pragma unroll
    for (int r = 0; r < 8; ++r) {
      int i = it * 16 + g * 8 + r;
      float q = (float)Qh[i * N + k];
      numacc[r] += acc[r] * q * v2v;
    }

    async_wait();
    __syncthreads();
  }

  // reduce over k: sum across the 16 lanes of each half-wave group
  #pragma unroll
  for (int r = 0; r < 8; ++r) {
    float v = numacc[r];
    for (int off = 1; off < 16; off <<= 1) v += __shfl_xor(v, off, 16);
    numacc[r] = v;
  }
  if (m == 0 && slot <= 64) {
    int i0 = it * 16 + g * 8;
    if (isNum) { for (int r = 0; r < 8; ++r) num[(i0 + r) * CC + slot] = numacc[r]; }
    else       { for (int r = 0; r < 8; ++r) den[i0 + r] = numacc[r]; }
  }
}

// ---------- stage 7: out = num / den ----------
__global__ void k_final(const float* __restrict__ num, const float* __restrict__ den,
                        float* __restrict__ out) {
  int idx = blockIdx.x * blockDim.x + threadIdx.x;
  if (idx >= N * CC) return;
  out[idx] = num[idx] / den[idx / CC];
}

// ---------------------------------------------------------------------------
extern "C" void kernel_launch(void* const* d_in, const int* in_sizes, int n_in,
                              void* d_out, int out_size, void* d_ws, size_t ws_size,
                              hipStream_t stream) {
  const float* x   = (const float*)d_in[0];   // [384,64]
  const float* w_w = (const float*)d_in[1];   // [512,64]
  float* out = (float*)d_out;                 // [384,64]

  char* base = (char*)d_ws;
  size_t off = 0;
  auto alloc = [&](size_t bytes) -> char* {
    char* p = base + off;
    off = (off + bytes + 255) & ~(size_t)255;
    return p;
  };

  f16*   xh  = (f16*)  alloc((size_t)N * CC * 2);
  f16*   wh  = (f16*)  alloc((size_t)8 * CC * CC * 2);
  f16*   w8h = (f16*)  alloc((size_t)N * 512 * 2);
  float* S   = (float*)alloc((size_t)3 * N * N * 4);
  float* m3  = (float*)alloc(3 * 4);
  f16*   Ph  = (f16*)  alloc((size_t)N * N * 2);
  f16*   Qh  = (f16*)  alloc((size_t)N * N * 2);
  f16*   Rth = (f16*)  alloc((size_t)N * N * 2);
  f16*   v1t = (f16*)  alloc((size_t)CC * N * 2);
  f16*   v2t = (f16*)  alloc((size_t)CC * N * 2);
  float* num = (float*)alloc((size_t)N * CC * 4);
  float* den = (float*)alloc((size_t)N * 4);

  {
    int n = N * CC + 8 * CC * CC;
    k_cvt<<<(n + 255) / 256, 256, 0, stream>>>(x, w_w, xh, wh);
  }
  k_proj<<<dim3(NT, 32), 32, 0, stream>>>(xh, wh, w8h);
  k_scores<<<dim3(NT, NT, 3), 32, 0, stream>>>(w8h, S);
  k_max3<<<3, 256, 0, stream>>>(S, m3);
  {
    int n = 3 * N * N;
    k_exp<<<(n + 255) / 256, 256, 0, stream>>>(S, m3, Ph, Qh, Rth);
  }
  {
    int n = 2 * CC * N;
    k_vt<<<(n + 255) / 256, 256, 0, stream>>>(w8h, v1t, v2t);
  }
  // 24 i-tiles x 9 slot-groups (8 waves each -> d=0..63 plus den)
  k_main<<<dim3(NT, 9), 256, 0, stream>>>(Ph, Qh, Rth, v1t, v2t, num, den);
  k_final<<<(N * CC + 255) / 256, 256, 0, stream>>>(num, den, out);
}
